// Head_67980742361700
// MI455X (gfx1250) — matrix operand order repaired and checked by reading, not compiled, not run on hardware
//
#include <hip/hip_runtime.h>

typedef _Float16 v16h __attribute__((ext_vector_type(16)));
typedef _Float16 v8h  __attribute__((ext_vector_type(8)));
typedef float    v8f  __attribute__((ext_vector_type(8)));

#define BATCH 8
#define TSEQ  2048
#define CDIM  2048
#define HSZ   128
#define NROW  (BATCH * TSEQ)   // 16384

// ---- optional CDNA5 async global->LDS path (guarded; falls back cleanly) ----
#if defined(__has_builtin)
#if __has_builtin(__builtin_amdgcn_global_load_async_to_lds_b128)
#define HAVE_ASYNC_LDS 1
#endif
#endif
#ifndef HAVE_ASYNC_LDS
#define HAVE_ASYNC_LDS 0
#endif

#if HAVE_ASYNC_LDS
typedef int v4i32 __attribute__((vector_size(16)));
typedef __attribute__((address_space(1))) v4i32* g_v4i_ptr;   // global src
typedef __attribute__((address_space(3))) v4i32* s_v4i_ptr;   // LDS dst
static __device__ __forceinline__ void async_cp16(const _Float16* g, _Float16* l) {
  __builtin_amdgcn_global_load_async_to_lds_b128((g_v4i_ptr)g, (s_v4i_ptr)l, 0, 0);
}
static __device__ __forceinline__ void wait_async0() {
#if __has_builtin(__builtin_amdgcn_s_wait_asynccnt)
  __builtin_amdgcn_s_wait_asynccnt(0);
#else
  asm volatile("s_wait_asynccnt 0" ::: "memory");
#endif
}
#endif

static __device__ __forceinline__ v16h cat16(v8h lo, v8h hi) {
  return __builtin_shufflevector(lo, hi, 0,1,2,3,4,5,6,7,8,9,10,11,12,13,14,15);
}

// ---------------------------------------------------------------------------
// Kernel 0: RoPE tables  cos/sin[t][i], i = 0..63 (pair index), theta=10000
// ---------------------------------------------------------------------------
__global__ void rope_tables_kernel(float* __restrict__ cosT, float* __restrict__ sinT) {
  int tid = blockIdx.x * blockDim.x + threadIdx.x;
  if (tid >= TSEQ * 64) return;
  int t = tid >> 6, i = tid & 63;
  float freq = __expf(-(float)(2 * i) * (9.210340371976184f / 128.0f));
  float ang = (float)t * freq;
  float s, c;
  __sincosf(ang, &s, &c);
  cosT[tid] = c;
  sinT[tid] = s;
}

// ---------------------------------------------------------------------------
// Kernel 1: weights f32 [C][HS] -> f16 transposed Wt[3][HS][C]  (0=Wq,1=Wk,2=Wv)
// ---------------------------------------------------------------------------
__global__ void wconv_kernel(const float* __restrict__ Wq, const float* __restrict__ Wk,
                             const float* __restrict__ Wv, _Float16* __restrict__ Wt) {
  int idx = blockIdx.x * blockDim.x + threadIdx.x;  // over CDIM*HSZ
  int z = blockIdx.y;
  const float* src = (z == 0) ? Wq : (z == 1) ? Wk : Wv;
  int c = idx >> 7, h = idx & 127;
  Wt[(size_t)z * HSZ * CDIM + (size_t)h * CDIM + c] = (_Float16)src[idx];
}

// ---------------------------------------------------------------------------
// Kernel 2: projection GEMM, 64x128 tile, double-buffered LDS, WMMA f16 16x16x32
//           epilogue: z=0 -> qr (RoPE), z=1 -> kr (RoPE), z=2 -> vt[b][d][t]
// ---------------------------------------------------------------------------
__global__ __launch_bounds__(256, 1)
void proj_rope_kernel(const float* __restrict__ x, const _Float16* __restrict__ Wt,
                      const float* __restrict__ cosT, const float* __restrict__ sinT,
                      _Float16* __restrict__ qr, _Float16* __restrict__ kr,
                      _Float16* __restrict__ vt) {
  __shared__ _Float16 As[2][64 * 40];    // x tile (f16, padded stride)
  __shared__ _Float16 Bs[2][128 * 40];   // Wt tile [n][k]
  __shared__ float    Cs[64 * 128];      // f32 staging for epilogue

  const int z = blockIdx.z;
  const int mbase = blockIdx.x * 64;
  const int tid = threadIdx.x;
  const int lane = tid & 31, wid = tid >> 5;
  const int mstrip = wid & 3, nhalf = wid >> 2;
  const int l15 = lane & 15;
  const int kb = (lane >> 4) * 8;     // A-frag K sub-offset
  const int hb = (lane >> 4) * 16;    // unused; B-frag also uses kb

  const _Float16* Wz = Wt + (size_t)z * HSZ * CDIM;

  const int arow = tid >> 2;          // 0..63
  const int acol = (tid & 3) * 8;     // 0,8,16,24
  const int brow = tid >> 1;          // 0..127
  const int bcol = (tid & 1) * 16;    // 0 or 16

  const float*    xp0 = x  + (size_t)(mbase + arow) * CDIM + acol;
  const _Float16* wp0 = Wz + (size_t)brow * CDIM + bcol;

  v8f acc[4] = {};

  float4 f0, f1;
#if !HAVE_ASYNC_LDS
  v8h wb0, wb1;
#endif

  // prologue: start tile 0
  f0 = *(const float4*)(xp0);
  f1 = *(const float4*)(xp0 + 4);
#if HAVE_ASYNC_LDS
  async_cp16(wp0,     &Bs[0][brow * 40 + bcol]);
  async_cp16(wp0 + 8, &Bs[0][brow * 40 + bcol + 8]);
#else
  wb0 = *(const v8h*)(wp0);
  wb1 = *(const v8h*)(wp0 + 8);
#endif

  const int NIT = CDIM / 32;
  for (int it = 0; it < NIT; ++it) {
    const int buf = it & 1;

    // commit staged tile into LDS buffer `buf`
    v8h ah;
    ah[0] = (_Float16)f0.x; ah[1] = (_Float16)f0.y; ah[2] = (_Float16)f0.z; ah[3] = (_Float16)f0.w;
    ah[4] = (_Float16)f1.x; ah[5] = (_Float16)f1.y; ah[6] = (_Float16)f1.z; ah[7] = (_Float16)f1.w;
    *(v8h*)&As[buf][arow * 40 + acol] = ah;
#if HAVE_ASYNC_LDS
    wait_async0();                       // B tile for `buf` landed in LDS
#else
    *(v8h*)&Bs[buf][brow * 40 + bcol] = wb0;
    *(v8h*)&Bs[buf][brow * 40 + bcol + 8] = wb1;
#endif
    __syncthreads();                     // single barrier per iteration

    // start next tile into buffer buf^1 (overlaps with WMMA below)
    if (it + 1 < NIT) {
      const float* xp = xp0 + (it + 1) * 32;
      f0 = *(const float4*)xp;
      f1 = *(const float4*)(xp + 4);
      const _Float16* wp = wp0 + (it + 1) * 32;
#if HAVE_ASYNC_LDS
      async_cp16(wp,     &Bs[buf ^ 1][brow * 40 + bcol]);
      async_cp16(wp + 8, &Bs[buf ^ 1][brow * 40 + bcol + 8]);
#else
      wb0 = *(const v8h*)wp;
      wb1 = *(const v8h*)(wp + 8);
#endif
    }

    // batch all fragment loads, then back-to-back WMMAs
    const int ar = mstrip * 16 + l15;
    v8h alo = *(const v8h*)&As[buf][ar * 40 + kb];
    v8h ahi = *(const v8h*)&As[buf][ar * 40 + 16 + kb];
    v8h bl[4], bh[4];
    #pragma unroll
    for (int nt = 0; nt < 4; ++nt) {
      int nr = nhalf * 64 + nt * 16 + l15;
      bl[nt] = *(const v8h*)&Bs[buf][nr * 40 + kb];
      bh[nt] = *(const v8h*)&Bs[buf][nr * 40 + 16 + kb];
    }
    v16h af = cat16(alo, ahi);
    #pragma unroll
    for (int nt = 0; nt < 4; ++nt) {
      acc[nt] = __builtin_amdgcn_wmma_f32_16x16x32_f16(false, af, false, cat16(bl[nt], bh[nt]),
                                                       (short)0, acc[nt], false, false);
    }
    // no trailing barrier: double buffering makes one barrier per iter sufficient
  }
  __syncthreads();   // protect Cs (reuse of LDS phase) vs stragglers

  // C tile -> LDS staging (VGPR j: rows j / j+8 across lane halves, col = lane&15)
  #pragma unroll
  for (int nt = 0; nt < 4; ++nt) {
    #pragma unroll
    for (int j = 0; j < 8; ++j) {
      int r = mstrip * 16 + j + ((lane >> 4) << 3);
      int ccol = nhalf * 64 + nt * 16 + l15;
      Cs[r * 128 + ccol] = acc[nt][j];
    }
  }
  __syncthreads();

  if (z < 2) {
    _Float16* dst = (z == 0) ? qr : kr;
    int row = tid >> 2;
    int pbase = (tid & 3) * 16;
    int grow = mbase + row;
    int tpos = grow & (TSEQ - 1);
    _Float16 buf16[32];
    #pragma unroll
    for (int p = 0; p < 16; ++p) {
      int pp = pbase + p;
      float x0 = Cs[row * 128 + 2 * pp];
      float x1 = Cs[row * 128 + 2 * pp + 1];
      float c = cosT[tpos * 64 + pp];
      float s = sinT[tpos * 64 + pp];
      buf16[2 * p]     = (_Float16)(x0 * c - x1 * s);
      buf16[2 * p + 1] = (_Float16)(x0 * s + x1 * c);
    }
    _Float16* op = dst + (size_t)grow * HSZ + 2 * pbase;
    #pragma unroll
    for (int v = 0; v < 4; ++v) *(v8h*)(op + v * 8) = *(v8h*)&buf16[v * 8];
  } else {
    int d = tid >> 1;
    int q0 = (tid & 1) * 32;
    int b = mbase >> 11;                 // TSEQ = 2048
    int tpos0 = (mbase & (TSEQ - 1)) + q0;
    _Float16 buf16[32];
    #pragma unroll
    for (int qq = 0; qq < 32; ++qq) buf16[qq] = (_Float16)Cs[(q0 + qq) * 128 + d];
    _Float16* op = vt + ((size_t)b * HSZ + d) * TSEQ + tpos0;
    #pragma unroll
    for (int v = 0; v < 4; ++v) *(v8h*)(op + v * 8) = *(v8h*)&buf16[v * 8];
  }
}

// ---------------------------------------------------------------------------
// Kernel 3: causal flash attention, 64 query rows/block, 4 waves x 16 rows,
//           32-key tiles, all matmuls via v_wmma_f32_16x16x32_f16.
// ---------------------------------------------------------------------------
__global__ __launch_bounds__(128, 1)
void attn_kernel(const _Float16* __restrict__ qr, const _Float16* __restrict__ kr,
                 const _Float16* __restrict__ vt, float* __restrict__ out) {
  __shared__ _Float16 Pbuf[4][16 * 32];   // per-wave P staging (C-layout -> A-frag)

  const int b = blockIdx.y;
  const int qtile = blockIdx.x * 64;
  const int lane = threadIdx.x & 31, wid = threadIdx.x >> 5;
  const int l15 = lane & 15;
  const int hl = lane >> 4;
  const int kb = hl * 8;      // A-frag K sub-offset
  const int hb = hl * 16;     // unused; B-frag also uses kb

  // resident Q fragments
  const int qrow_a = qtile + wid * 16 + l15;
  const _Float16* qp = qr + ((size_t)b * TSEQ + qrow_a) * HSZ;
  v16h qf[4];
  #pragma unroll
  for (int kc = 0; kc < 4; ++kc) {
    v8h lo = *(const v8h*)(qp + kc * 32 + kb);
    v8h hi = *(const v8h*)(qp + kc * 32 + 16 + kb);
    qf[kc] = cat16(lo, hi);
  }

  const _Float16* krb = kr + (size_t)b * TSEQ * HSZ;
  const _Float16* vtb = vt + (size_t)b * HSZ * TSEQ;

  v8f acc[8] = {};
  float mrow[8], lrow[8];
  #pragma unroll
  for (int j = 0; j < 8; ++j) { mrow[j] = -1e30f; lrow[j] = 0.0f; }

  const float scale = 0.08838834764831845f;   // 1/sqrt(128)

  for (int k0 = 0; k0 < qtile + 64; k0 += 32) {
    // ---- S = Q @ K^T, two 16-key subtiles; batch loads, then WMMA chains ----
    v8f s0 = {}, s1 = {};
    const _Float16* kp0 = krb + (size_t)(k0 + l15) * HSZ;
    const _Float16* kp1 = kp0 + (size_t)16 * HSZ;
    {
      v16h kf[4];
      #pragma unroll
      for (int kc = 0; kc < 4; ++kc) {
        v8h lo = *(const v8h*)(kp0 + kc * 32 + kb);
        v8h hi = *(const v8h*)(kp0 + kc * 32 + 16 + kb);
        kf[kc] = cat16(lo, hi);
      }
      #pragma unroll
      for (int kc = 0; kc < 4; ++kc)
        s0 = __builtin_amdgcn_wmma_f32_16x16x32_f16(false, qf[kc], false, kf[kc],
                                                    (short)0, s0, false, false);
      #pragma unroll
      for (int kc = 0; kc < 4; ++kc) {
        v8h lo = *(const v8h*)(kp1 + kc * 32 + kb);
        v8h hi = *(const v8h*)(kp1 + kc * 32 + 16 + kb);
        kf[kc] = cat16(lo, hi);
      }
      #pragma unroll
      for (int kc = 0; kc < 4; ++kc)
        s1 = __builtin_amdgcn_wmma_f32_16x16x32_f16(false, qf[kc], false, kf[kc],
                                                    (short)0, s1, false, false);
    }

    // ---- online softmax (cross-lane reduce within 16-lane C-tile halves) ----
    int key0 = k0 + l15, key1 = key0 + 16;
    #pragma unroll
    for (int j = 0; j < 8; ++j) {
      int row = qtile + wid * 16 + j + hl * 8;
      float v0 = (key0 <= row) ? s0[j] * scale : -1e30f;
      float v1 = (key1 <= row) ? s1[j] * scale : -1e30f;
      float mx = fmaxf(v0, v1);
      mx = fmaxf(mx, __shfl_xor(mx, 1, 32));
      mx = fmaxf(mx, __shfl_xor(mx, 2, 32));
      mx = fmaxf(mx, __shfl_xor(mx, 4, 32));
      mx = fmaxf(mx, __shfl_xor(mx, 8, 32));
      float mnew = fmaxf(mrow[j], mx);
      float alpha = __expf(mrow[j] - mnew);
      float p0 = __expf(v0 - mnew);
      float p1 = __expf(v1 - mnew);
      float rs = p0 + p1;
      rs += __shfl_xor(rs, 1, 32);
      rs += __shfl_xor(rs, 2, 32);
      rs += __shfl_xor(rs, 4, 32);
      rs += __shfl_xor(rs, 8, 32);
      lrow[j] = lrow[j] * alpha + rs;
      mrow[j] = mnew;
      #pragma unroll
      for (int d = 0; d < 8; ++d) acc[d][j] *= alpha;
      int r = j + hl * 8;
      Pbuf[wid][r * 32 + l15]      = (_Float16)p0;
      Pbuf[wid][r * 32 + 16 + l15] = (_Float16)p1;
    }
    // wave-private LDS region; DS ops in-order per wave — compiler barrier only
    asm volatile("" ::: "memory");

    // P as A-fragment (16 x 32 keys)
    v8h plo = *(const v8h*)&Pbuf[wid][l15 * 32 + kb];
    v8h phi = *(const v8h*)&Pbuf[wid][l15 * 32 + 16 + kb];
    v16h pf = cat16(plo, phi);

    // ---- acc += P @ V : batch 4 fragments, 4 WMMAs, twice ----
    #pragma unroll
    for (int g = 0; g < 2; ++g) {
      v16h vf[4];
      #pragma unroll
      for (int d = 0; d < 4; ++d) {
        const _Float16* vp = vtb + (size_t)((g * 4 + d) * 16 + l15) * TSEQ + k0;
        v8h lo = *(const v8h*)(vp + kb);
        v8h hi = *(const v8h*)(vp + 16 + kb);
        vf[d] = cat16(lo, hi);
      }
      #pragma unroll
      for (int d = 0; d < 4; ++d)
        acc[g * 4 + d] = __builtin_amdgcn_wmma_f32_16x16x32_f16(false, pf, false, vf[d],
                                                                (short)0, acc[g * 4 + d],
                                                                false, false);
    }
  }

  // epilogue: one reciprocal per row, then normalize + store f32
  float invl[8];
  #pragma unroll
  for (int j = 0; j < 8; ++j) invl[j] = 1.0f / lrow[j];
  #pragma unroll
  for (int d = 0; d < 8; ++d) {
    #pragma unroll
    for (int j = 0; j < 8; ++j) {
      int row = qtile + wid * 16 + j + hl * 8;
      out[((size_t)b * TSEQ + row) * HSZ + d * 16 + l15] = acc[d][j] * invl[j];
    }
  }
}

// ---------------------------------------------------------------------------
extern "C" void kernel_launch(void* const* d_in, const int* in_sizes, int n_in,
                              void* d_out, int out_size, void* d_ws, size_t ws_size,
                              hipStream_t stream) {
  const float* x  = (const float*)d_in[0];
  const float* Wk = (const float*)d_in[1];
  const float* Wq = (const float*)d_in[2];
  const float* Wv = (const float*)d_in[3];
  float* out = (float*)d_out;

  // workspace layout (~14.5 MB total)
  float* cosT = (float*)d_ws;                               // T*64 f32
  float* sinT = cosT + TSEQ * 64;                           // T*64 f32
  _Float16* Wt  = (_Float16*)(sinT + TSEQ * 64);            // 3*128*2048 f16
  _Float16* qrb = Wt + (size_t)3 * HSZ * CDIM;              // [B*T][128] f16
  _Float16* krb = qrb + (size_t)NROW * HSZ;                 // [B*T][128] f16
  _Float16* vtb = krb + (size_t)NROW * HSZ;                 // [B][128][T] f16

  rope_tables_kernel<<<dim3((TSEQ * 64) / 256), 256, 0, stream>>>(cosT, sinT);
  wconv_kernel<<<dim3((CDIM * HSZ) / 256, 3), 256, 0, stream>>>(Wq, Wk, Wv, Wt);
  proj_rope_kernel<<<dim3(NROW / 64, 1, 3), 256, 0, stream>>>(x, Wt, cosT, sinT, qrb, krb, vtb);
  attn_kernel<<<dim3(TSEQ / 64, BATCH), 128, 0, stream>>>(qrb, krb, vtb, out);
}
